// Detector_66288525246532
// MI455X (gfx1250) — matrix operand order repaired and checked by reading, not compiled, not run on hardware
//
#include <hip/hip_runtime.h>
#include <hip/hip_bf16.h>
#include <stdint.h>

// ---------------------------------------------------------------------------
// MI455X (gfx1250) FPN+RPN detector head.
// All convs run as implicit GEMMs on v_wmma_f32_16x16x32_bf16 (wave32).
// Activations: NHWC bf16.  Weights: [co][tap*Cin+ci] bf16.
// A-frag: lane<16 -> K 0..15, lane>=16 -> K 16..31  (2x 16B loads)
// B-frag: lane<16 -> K k0..k0+15 @ pixel=lane, lane>=16 -> K k0+16..k0+31 (1x 32B load)
// C/D   : VGPR r, lanes0-15 -> co=cb+r, lanes16-31 -> co=cb+8+r, pixel=lane&15
// Loop structure: tap-outer (bounds/addresses hoisted), channel-inner
// (loads + wmma + pointer bumps only).
// ---------------------------------------------------------------------------

typedef __attribute__((ext_vector_type(8)))  float  v8f;
typedef __attribute__((ext_vector_type(16))) __bf16 v16bf;
typedef __attribute__((ext_vector_type(8)))  __bf16 v8bf;

union V16U { v16bf v; v8bf h[2]; };

// ----------------------------- layout kernels ------------------------------

__global__ void k_nchw_to_nhwc_bf16(const float* __restrict__ in, __bf16* __restrict__ out,
                                    int C, int HW)
{
  long idx = (long)blockIdx.x * blockDim.x + threadIdx.x;
  long tot = (long)HW * C;
  if (idx >= tot) return;
  int  n = blockIdx.z;
  int  c = (int)(idx % C);
  long p = idx / C;
  out[(size_t)n * tot + idx] = (__bf16)in[((size_t)n * C + c) * HW + p];
}

__global__ void k_repack_1x1(const float* __restrict__ w, __bf16* __restrict__ o, int total)
{
  int i = blockIdx.x * blockDim.x + threadIdx.x;
  if (i < total) o[i] = (__bf16)w[i];
}

// OIHW [256][256][3][3] -> [co][tap*256+ci]
__global__ void k_repack_3x3(const float* __restrict__ w, __bf16* __restrict__ o)
{
  int i = blockIdx.x * blockDim.x + threadIdx.x;
  if (i >= 256 * 2304) return;
  int co = i / 2304, r = i % 2304;
  int tap = r / 256, ci = r % 256;
  o[i] = (__bf16)w[(co * 256 + ci) * 9 + tap];
}

// [15][256] -> [ci][16] (co padded to 16 with zeros)
__global__ void k_repack_rpn2(const float* __restrict__ w, __bf16* __restrict__ o)
{
  int i = blockIdx.x * blockDim.x + threadIdx.x;
  if (i >= 256 * 16) return;
  int ci = i >> 4, c = i & 15;
  o[i] = (c < 15) ? (__bf16)w[c * 256 + ci] : (__bf16)0.f;
}

// P4[n][y][x][c] = P3[n][2y][2x][c]
__global__ void k_subsample2(const __bf16* __restrict__ in, __bf16* __restrict__ out,
                             int H, int W, int Ho, int Wo)
{
  int n = blockIdx.z;
  int i = blockIdx.x * blockDim.x + threadIdx.x;
  int tot = Ho * Wo * 256;
  if (i >= tot) return;
  int c = i & 255, p = i >> 8;
  int y = p / Wo, x = p % Wo;
  out[(size_t)n * tot + i] = in[((size_t)n * H * W + (2 * y) * W + (2 * x)) * 256 + c];
}

// ------------------------- WMMA implicit-GEMM conv -------------------------
// y[n,p,co] = sum_{tap,ci} wt[co][tap*Cin+ci] * x[n, p+shift(tap), ci]  (+bias)
// optional: += up2(skip) (skip is H/2 x W/2 x 256), optional ReLU.
__global__ void __launch_bounds__(256)
k_conv_wmma(const __bf16* __restrict__ x, const __bf16* __restrict__ wt,
            const float* __restrict__ bias, const __bf16* __restrict__ skip,
            __bf16* __restrict__ y,
            int H, int W, int Cin, int k3, int relu)
{
  const int HW   = H * W;
  const int lane = threadIdx.x & 31;
  const int wv   = threadIdx.x >> 5;
  const int cb   = (blockIdx.y * 8 + wv) * 16;   // Cout tile base (Cout = 256)
  const int pb   = blockIdx.x * 64;              // pixel tile base (4 x 16)
  const int n    = blockIdx.z;
  const int col  = lane & 15;
  const int hi16 = lane >> 4;
  const int Ktot = k3 ? Cin * 9 : Cin;
  const int ntap = k3 ? 9 : 1;

  int  ph[4], pw[4];
  bool pv[4];
#pragma unroll
  for (int t = 0; t < 4; ++t) {
    int p = pb + t * 16 + col;
    pv[t] = (p < HW);
    int pp = pv[t] ? p : 0;
    ph[t] = pp / W;
    pw[t] = pp % W;
  }

  const __bf16* wrow = wt + (size_t)(cb + col) * Ktot + hi16 * 16;
  V16U zb;
#pragma unroll
  for (int i = 0; i < 16; ++i) zb.v[i] = (__bf16)0.f;

  v8f acc[4];
#pragma unroll
  for (int t = 0; t < 4; ++t)
#pragma unroll
    for (int i = 0; i < 8; ++i) acc[t][i] = 0.f;

  const __bf16* arow = wrow;
  for (int tap = 0; tap < ntap; ++tap) {
    // hoisted per-tap: shift, validity, and B base pointers
    const int dy = k3 ? (tap / 3 - 1) : 0;
    const int dx = k3 ? (tap % 3 - 1) : 0;
    const __bf16* bptr[4];
    bool          bval[4];
#pragma unroll
    for (int t = 0; t < 4; ++t) {
      int hh = ph[t] + dy, ww = pw[t] + dx;
      bval[t] = pv[t] && hh >= 0 && hh < H && ww >= 0 && ww < W;
      int ppix = bval[t] ? (hh * W + ww) : 0;
      bptr[t] = x + ((size_t)n * HW + ppix) * Cin + hi16 * 16;
    }
    // channel-inner hot loop: loads + wmma + pointer bumps only
#pragma unroll 2
    for (int c0 = 0; c0 < Cin; c0 += 32) {
      V16U a;
      a.h[0] = *(const v8bf*)(arow + c0);
      a.h[1] = *(const v8bf*)(arow + c0 + 8);
      __builtin_prefetch((const void*)(arow + c0 + 32), 0, 0);  // global_prefetch_b8
#pragma unroll
      for (int t = 0; t < 4; ++t) {
        V16U b;
        if (bval[t])
          b.v = *(const v16bf*)(bptr[t] + c0);
        else
          b = zb;
        acc[t] = __builtin_amdgcn_wmma_f32_16x16x32_bf16(
            false, a.v, false, b.v, (short)0, acc[t], false, false);
      }
    }
    arow += Cin;
  }

  // epilogue: bias (+ up2 skip) (+ relu), store bf16 NHWC
  const int cofs = cb + hi16 * 8;
  float bv[8];
#pragma unroll
  for (int r = 0; r < 8; ++r) bv[r] = bias[cofs + r];

#pragma unroll
  for (int t = 0; t < 4; ++t) {
    if (!pv[t]) continue;
    v8f o = acc[t];
#pragma unroll
    for (int r = 0; r < 8; ++r) o[r] += bv[r];
    if (skip) {
      int Hs = H >> 1, Ws = W >> 1;
      v8bf s = *(const v8bf*)(skip +
          ((size_t)n * Hs * Ws + (size_t)(ph[t] >> 1) * Ws + (pw[t] >> 1)) * 256 + cofs);
#pragma unroll
      for (int r = 0; r < 8; ++r) o[r] += (float)s[r];
    }
    if (relu) {
#pragma unroll
      for (int r = 0; r < 8; ++r) o[r] = o[r] > 0.f ? o[r] : 0.f;
    }
    v8bf ob;
#pragma unroll
    for (int r = 0; r < 8; ++r) ob[r] = (__bf16)o[r];
    int p = pb + t * 16 + col;
    *(v8bf*)(y + ((size_t)n * HW + p) * 256 + cofs) = ob;
  }
}

// --------------- RPN 1x1 conv 256->15 + transpose scatter ------------------
// out[n, rowBase + p*3 + a, c] = sum_ci h[n,p,ci]*w2[ci][a*5+c] + b2[a*5+c]
__global__ void k_rpn_out(const __bf16* __restrict__ h, const __bf16* __restrict__ w2,
                          const float* __restrict__ b2, float* __restrict__ out,
                          int HW, long rowBase, long Rtot)
{
  int n = blockIdx.z;
  int p = blockIdx.x * blockDim.x + threadIdx.x;
  if (p >= HW) return;
  const __bf16* hp = h + ((size_t)n * HW + p) * 256;
  float acc[16];
#pragma unroll
  for (int j = 0; j < 16; ++j) acc[j] = 0.f;
  for (int ci = 0; ci < 256; ci += 8) {
    v8bf hv = *(const v8bf*)(hp + ci);
#pragma unroll
    for (int u = 0; u < 8; ++u) {
      float hf = (float)hv[u];
      v16bf wr = *(const v16bf*)(w2 + (ci + u) * 16);
#pragma unroll
      for (int j = 0; j < 16; ++j) acc[j] += hf * (float)wr[j];
    }
  }
  float* o = out + ((size_t)n * Rtot + rowBase + (long)p * 3) * 5;
#pragma unroll
  for (int j = 0; j < 15; ++j) o[j] = acc[j] + b2[j];
}

// ------------------------------- launcher ----------------------------------

extern "C" void kernel_launch(void* const* d_in, const int* in_sizes, int n_in,
                              void* d_out, int out_size, void* d_ws, size_t ws_size,
                              hipStream_t stream)
{
  (void)n_in; (void)out_size; (void)ws_size;
  const int Hs[4] = {200, 100, 50, 25};
  const int Ci[4] = {256, 512, 1024, 2048};
  const int B = in_sizes[0] / (256 * 200 * 200);

  const float* c[4]    = {(const float*)d_in[0], (const float*)d_in[1],
                          (const float*)d_in[2], (const float*)d_in[3]};
  const float* latw[4] = {(const float*)d_in[4], (const float*)d_in[6],
                          (const float*)d_in[8], (const float*)d_in[10]};
  const float* latb[4] = {(const float*)d_in[5], (const float*)d_in[7],
                          (const float*)d_in[9], (const float*)d_in[11]};
  const float* trw[4]  = {(const float*)d_in[12], (const float*)d_in[14],
                          (const float*)d_in[16], (const float*)d_in[18]};
  const float* trb[4]  = {(const float*)d_in[13], (const float*)d_in[15],
                          (const float*)d_in[17], (const float*)d_in[19]};
  const float* rw1 = (const float*)d_in[20];
  const float* rb1 = (const float*)d_in[21];
  const float* rw2 = (const float*)d_in[22];
  const float* rb2 = (const float*)d_in[23];

  uint8_t* base = (uint8_t*)d_ws;
  size_t   off  = 0;
  auto alloc = [&](size_t bytes) -> void* {
    void* p = base + off;
    off = (off + bytes + 255) & ~(size_t)255;
    return p;
  };

  __bf16 *xin[4], *M[4], *P[5], *Hb;
  __bf16 *wlat[4], *wtr[4], *wr1, *wr2;
  for (int i = 0; i < 4; ++i) xin[i] = (__bf16*)alloc((size_t)B * Hs[i] * Hs[i] * Ci[i] * 2);
  for (int i = 0; i < 4; ++i) M[i]   = (__bf16*)alloc((size_t)B * Hs[i] * Hs[i] * 256 * 2);
  for (int i = 0; i < 4; ++i) P[i]   = (__bf16*)alloc((size_t)B * Hs[i] * Hs[i] * 256 * 2);
  P[4] = (__bf16*)alloc((size_t)B * 13 * 13 * 256 * 2);
  Hb   = (__bf16*)alloc((size_t)B * 200 * 200 * 256 * 2);
  for (int i = 0; i < 4; ++i) wlat[i] = (__bf16*)alloc((size_t)256 * Ci[i] * 2);
  for (int i = 0; i < 4; ++i) wtr[i]  = (__bf16*)alloc((size_t)256 * 2304 * 2);
  wr1 = (__bf16*)alloc((size_t)256 * 2304 * 2);
  wr2 = (__bf16*)alloc((size_t)256 * 16 * 2);

  // fp32 NCHW -> bf16 NHWC
  for (int i = 0; i < 4; ++i) {
    long tot = (long)Hs[i] * Hs[i] * Ci[i];
    dim3 g((unsigned)((tot + 255) / 256), 1, B);
    k_nchw_to_nhwc_bf16<<<g, 256, 0, stream>>>(c[i], xin[i], Ci[i], Hs[i] * Hs[i]);
  }
  // weight repacks
  for (int i = 0; i < 4; ++i) {
    int tot = 256 * Ci[i];
    k_repack_1x1<<<(tot + 255) / 256, 256, 0, stream>>>(latw[i], wlat[i], tot);
    k_repack_3x3<<<(256 * 2304 + 255) / 256, 256, 0, stream>>>(trw[i], wtr[i]);
  }
  k_repack_3x3<<<(256 * 2304 + 255) / 256, 256, 0, stream>>>(rw1, wr1);
  k_repack_rpn2<<<(256 * 16 + 255) / 256, 256, 0, stream>>>(rw2, wr2);

  auto conv = [&](const __bf16* x, const __bf16* wt, const float* bias,
                  const __bf16* skip, __bf16* y, int H, int W, int Cin, int k3, int relu) {
    dim3 g((unsigned)((H * W + 63) / 64), 2, B);
    k_conv_wmma<<<g, 256, 0, stream>>>(x, wt, bias, skip, y, H, W, Cin, k3, relu);
  };

  // FPN top-down (lateral 1x1 + fused up2-add)
  conv(xin[3], wlat[3], latb[3], nullptr, M[3], 25, 25, 2048, 0, 0);
  conv(xin[2], wlat[2], latb[2], M[3],    M[2], 50, 50, 1024, 0, 0);
  conv(xin[1], wlat[1], latb[1], M[2],    M[1], 100, 100, 512, 0, 0);
  conv(xin[0], wlat[0], latb[0], M[1],    M[0], 200, 200, 256, 0, 0);

  // 3x3 transforms
  for (int i = 0; i < 4; ++i)
    conv(M[i], wtr[i], trb[i], nullptr, P[i], Hs[i], Hs[i], 256, 1, 0);

  // extra level: stride-2 subsample of P3
  {
    int tot = 13 * 13 * 256;
    dim3 g((unsigned)((tot + 255) / 256), 1, B);
    k_subsample2<<<g, 256, 0, stream>>>(P[3], P[4], 25, 25, 13, 13);
  }

  // RPN head per level
  const int Hl[5] = {200, 100, 50, 25, 13};
  long bases[5], rsum = 0;
  for (int l = 0; l < 5; ++l) { bases[l] = rsum; rsum += (long)Hl[l] * Hl[l] * 3; }
  const long Rtot = rsum;  // 159882
  for (int l = 0; l < 5; ++l) {
    conv(P[l], wr1, rb1, nullptr, Hb, Hl[l], Hl[l], 256, 1, 1);  // 3x3 + ReLU
    int HW = Hl[l] * Hl[l];
    dim3 g((unsigned)((HW + 255) / 256), 1, B);
    k_rpn_out<<<g, 256, 0, stream>>>(Hb, wr2, rb2, (float*)d_out, HW, bases[l], Rtot);
  }
}
